// ESN_52587579572231
// MI455X (gfx1250) — compile-verified, matrix operand order checked by reading, not run on hardware
//
#include <hip/hip_runtime.h>
#include <cstdint>

typedef unsigned short u16;
typedef __attribute__((ext_vector_type(16))) __bf16 v16bf;
typedef __attribute__((ext_vector_type(8)))  float  v8f;

#define ESN_B 64
#define ESN_T 512
#define ESN_I 128
#define ESN_H 2048

// LDS tile strides (in bf16 elements); padded, 16B aligned.
#define HS_STRIDE (ESN_H + 32)   // 2080
#define XS_STRIDE (ESN_I + 8)    // 136
#define SMEM_BYTES (16 * HS_STRIDE * 2 + 16 * XS_STRIDE * 2)

union Frag16 {
    uint4  u[2];
    v16bf  v;
};

__device__ __forceinline__ u16 f2bf(float f) {
    unsigned u = __float_as_uint(f);
    unsigned r = (u + 0x7FFFu + ((u >> 16) & 1u)) >> 16;  // RNE
    return (u16)r;
}

// Branch-free tanh on the TRANS pipe: v_exp_f32 + v_rcp_f32, no exec games.
__device__ __forceinline__ float fast_tanh(float x) {
    const float ax = __builtin_fabsf(x);
    const float e  = __expf(-2.0f * ax);                      // exp(-2|x|) in (0,1]
    const float r  = (1.0f - e) * __builtin_amdgcn_rcpf(1.0f + e);
    return __builtin_copysignf(r, x);
}

__device__ __forceinline__ v16bf load_frag(const u16* p) {
    Frag16 f;
    f.u[0] = *(const uint4*)(p);
    f.u[1] = *(const uint4*)(p + 16);
    return f.v;
}

__device__ __forceinline__ v8f wmma_bf16(v16bf a, v16bf b, v8f c) {
    return __builtin_amdgcn_wmma_f32_16x16x32_bf16(
        false, a, false, b, (short)0, c, false, false);
}

// Async copy of one 16-byte chunk: global -> LDS, tracked by ASYNCcnt.
__device__ __forceinline__ void async_cp16(u16* lds_dst, const u16* gsrc) {
    const unsigned           l = (unsigned)(uintptr_t)lds_dst;          // LDS byte addr
    const unsigned long long g = (unsigned long long)(uintptr_t)gsrc;   // 64-bit global
    asm volatile("global_load_async_to_lds_b128 %0, %1, off"
                 :: "v"(l), "v"(g) : "memory");
}

// ---------------------------------------------------------------------------
// Init: convert inputs to bf16 in workspace, zero h ping-pong + barrier count.
// ---------------------------------------------------------------------------
__global__ void esn_init_kernel(const float* __restrict__ x,
                                const float* __restrict__ w_in,
                                const float* __restrict__ w_res,
                                u16* __restrict__ xb,
                                u16* __restrict__ wib,
                                u16* __restrict__ wrb,
                                u16* __restrict__ h0,
                                u16* __restrict__ h1,
                                unsigned* __restrict__ cnt) {
    const size_t tid    = (size_t)blockIdx.x * blockDim.x + threadIdx.x;
    const size_t stride = (size_t)gridDim.x * blockDim.x;
    for (size_t i = tid; i < (size_t)ESN_B * ESN_T * ESN_I; i += stride) xb[i]  = f2bf(x[i]);
    for (size_t i = tid; i < (size_t)ESN_H * ESN_I;         i += stride) wib[i] = f2bf(w_in[i]);
    for (size_t i = tid; i < (size_t)ESN_H * ESN_H;         i += stride) wrb[i] = f2bf(w_res[i]);
    for (size_t i = tid; i < (size_t)ESN_B * ESN_H;         i += stride) { h0[i] = 0; h1[i] = 0; }
    if (tid == 0) *cnt = 0u;
}

// ---------------------------------------------------------------------------
// Persistent recurrent kernel: 64 WGs x 256 threads (8 wave32).
//   WG (mtile, ntile): batch rows [mtile*16, +16), out cols [ntile*128, +128).
//   wave w owns the 16x16 subtile at col ntile*128 + w*16.
// ---------------------------------------------------------------------------
__global__ void __launch_bounds__(256, 1)
esn_recurrent_kernel(const u16* __restrict__ xb,     // [B,T,I] bf16
                     const u16* __restrict__ wib,    // [H,I]   bf16
                     const u16* __restrict__ wrb,    // [H,H]   bf16
                     u16* __restrict__ h0,           // [B,H]   bf16 ping
                     u16* __restrict__ h1,           // [B,H]   bf16 pong
                     unsigned* __restrict__ cnt,     // barrier counter
                     float* __restrict__ out) {      // [B,H]   f32
    extern __shared__ u16 smem[];
    u16* hs = smem;                          // 16 x HS_STRIDE bf16
    u16* xs = smem + 16 * HS_STRIDE;         // 16 x XS_STRIDE bf16

    const int tid   = threadIdx.x;
    const int lane  = tid & 31;
    const int wave  = tid >> 5;              // 0..7
    const int mtile = blockIdx.x >> 4;       // 0..3
    const int ntile = blockIdx.x & 15;       // 0..15
    const int Bbase = mtile * 16;
    const int n0    = ntile * 128 + wave * 16;

    // WMMA 16-bit A/B operand addressing (16x32): lane -> row = lane%16,
    // two contiguous 16-byte chunks at K = (lane/16)*8 and +16.
    const int frow = lane & 15;
    const int ksub = (lane >> 4) << 3;       // 0 or 8
    const int ncol = n0 + (lane & 15);
    const int moff = (lane >> 4) << 3;       // C/D row offset: 0 or 8

    // Per-lane row bases (hoisted out of the time loop).
    const u16* hrow  = hs + frow * HS_STRIDE + ksub;        // A (recurrent) in LDS
    const u16* xrow  = xs + frow * XS_STRIDE + ksub;        // A (input) in LDS
    const u16* wrow  = wrb + (size_t)ncol * ESN_H + ksub;   // B (recurrent), L2-resident
    const u16* wirow = wib + (size_t)ncol * ESN_I + ksub;   // B (input)

    const unsigned nblocks = gridDim.x;

    u16* hcur  = h0;
    u16* hnext = h1;

    for (int t = 0; t < ESN_T; ++t) {
        // ---- async-stage h_{t-1} tile (16x2048 bf16) and x_t tile (16x128)
        for (int idx = tid; idx < 16 * (ESN_H / 8); idx += 256) {   // 16 chunks/thread
            const int r = idx >> 8;                  // ESN_H/8 == 256 chunks per row
            const int c = idx & 255;
            async_cp16(hs + r * HS_STRIDE + c * 8,
                       hcur + (size_t)(Bbase + r) * ESN_H + c * 8);
        }
        {                                            // 256 chunks == 1/thread
            const int r = tid >> 4;                  // ESN_I/8 == 16 chunks per row
            const int c = tid & 15;
            async_cp16(xs + r * XS_STRIDE + c * 8,
                       xb + ((size_t)(Bbase + r) * ESN_T + t) * ESN_I + c * 8);
        }
        asm volatile("s_wait_asynccnt 0x0" ::: "memory");
        __syncthreads();

        // Four independent accumulator chains keep the XDL pipe full.
        v8f acc0 = {}, acc1 = {}, acc2 = {}, acc3 = {};

        // ---- input contribution: x_t (16xI) x W_in^T  (K = 128, 4 chunks)
        acc0 = wmma_bf16(load_frag(xrow +  0), load_frag(wirow +  0), acc0);
        acc1 = wmma_bf16(load_frag(xrow + 32), load_frag(wirow + 32), acc1);
        acc2 = wmma_bf16(load_frag(xrow + 64), load_frag(wirow + 64), acc2);
        acc3 = wmma_bf16(load_frag(xrow + 96), load_frag(wirow + 96), acc3);

        // ---- recurrent contribution: h_{t-1} (16xH) x W_res^T  (K = 2048)
        #pragma unroll 2
        for (int k0 = 0; k0 < ESN_H; k0 += 128) {
            acc0 = wmma_bf16(load_frag(hrow + k0),      load_frag(wrow + k0),      acc0);
            acc1 = wmma_bf16(load_frag(hrow + k0 + 32), load_frag(wrow + k0 + 32), acc1);
            acc2 = wmma_bf16(load_frag(hrow + k0 + 64), load_frag(wrow + k0 + 64), acc2);
            acc3 = wmma_bf16(load_frag(hrow + k0 + 96), load_frag(wrow + k0 + 96), acc3);
        }

        const v8f acc = (acc0 + acc1) + (acc2 + acc3);

        // ---- activation + store h_t (bf16); final step also writes f32 output
        #pragma unroll
        for (int r = 0; r < 8; ++r) {
            const float v = fast_tanh(acc[r]);
            const size_t o = (size_t)(Bbase + r + moff) * ESN_H + ncol;
            hnext[o] = f2bf(v);
            if (t == ESN_T - 1) out[o] = v;
        }

        // ---- device-wide step barrier (monotonic arrival counter)
        __threadfence();          // release this WG's h_t stores
        __syncthreads();
        if (tid == 0) {
            __hip_atomic_fetch_add(cnt, 1u, __ATOMIC_RELEASE, __HIP_MEMORY_SCOPE_AGENT);
            const unsigned target = (unsigned)(t + 1) * nblocks;
            while (__hip_atomic_load(cnt, __ATOMIC_ACQUIRE, __HIP_MEMORY_SCOPE_AGENT) < target) {
                __builtin_amdgcn_s_sleep(2);
            }
        }
        __syncthreads();
        __threadfence();          // acquire other WGs' h_t stores

        u16* tmp = hcur; hcur = hnext; hnext = tmp;
    }
}

// ---------------------------------------------------------------------------
extern "C" void kernel_launch(void* const* d_in, const int* in_sizes, int n_in,
                              void* d_out, int out_size, void* d_ws, size_t ws_size,
                              hipStream_t stream) {
    (void)in_sizes; (void)n_in; (void)out_size; (void)ws_size;

    const float* x     = (const float*)d_in[0];   // [B,T,I]
    const float* w_in  = (const float*)d_in[1];   // [H,I]
    const float* w_res = (const float*)d_in[2];   // [H,H]
    float* out         = (float*)d_out;           // [B,H]

    char* ws = (char*)d_ws;
    size_t off = 0;
    unsigned* cnt = (unsigned*)(ws + off); off += 256;
    u16* h0  = (u16*)(ws + off); off += (size_t)ESN_B * ESN_H * 2;
    u16* h1  = (u16*)(ws + off); off += (size_t)ESN_B * ESN_H * 2;
    u16* wrb = (u16*)(ws + off); off += (size_t)ESN_H * ESN_H * 2;
    u16* wib = (u16*)(ws + off); off += (size_t)ESN_H * ESN_I * 2;
    u16* xb  = (u16*)(ws + off); off += (size_t)ESN_B * ESN_T * ESN_I * 2;

    esn_init_kernel<<<1024, 256, 0, stream>>>(x, w_in, w_res, xb, wib, wrb, h0, h1, cnt);

    esn_recurrent_kernel<<<64, 256, SMEM_BYTES, stream>>>(xb, wib, wrb, h0, h1, cnt, out);
}
